// VisionMambaEncoder_52458730553554
// MI455X (gfx1250) — matrix operand order repaired and checked by reading, not compiled, not run on hardware
//
#include <hip/hip_runtime.h>
#include <cstdint>

// ---------------- problem constants ----------------
#define TOK   8192          // BATCH(32) * NP(256)
#define DIMC  768
#define DIC   1536
#define DSC   16
#define SEQ   256
#define NLAY  12
#define KANK  6144          // 768 * 8 (GS=5 padded to 8 per channel)

// ---------------- GEMM tiling ----------------
#define BM 128
#define BN 64
#define BK 32
#define BKP 40              // LDS row stride in ushorts: 80B, 16B-aligned, conflict-free

typedef __attribute__((ext_vector_type(16))) __bf16        v16bf;
typedef __attribute__((ext_vector_type(8)))  float         v8f;
typedef __attribute__((ext_vector_type(4)))  float         f32x4;
typedef __attribute__((ext_vector_type(4)))  unsigned int  u32x4;
typedef __attribute__((ext_vector_type(8)))  int           i32x8;
typedef __attribute__((ext_vector_type(4)))  int           i32x4;

enum { AMODE_PLAIN = 0, AMODE_IM2COL = 1, AMODE_KAN = 2 };
enum { EPI_NONE = 0, EPI_BIAS = 1, EPI_BIAS_POS = 2, EPI_SOFTPLUS_BIAS = 3, EPI_ADD2 = 4 };

#if defined(__has_builtin)
#  if __has_builtin(__builtin_amdgcn_tensor_load_to_lds) && __has_builtin(__builtin_amdgcn_s_wait_tensorcnt)
#    define HAVE_TDM 1
#  else
#    define HAVE_TDM 0
#  endif
#else
#  define HAVE_TDM 0
#endif

__device__ __forceinline__ unsigned int f2bf1(float f) {
  union { float f; unsigned int u; } v; v.f = f;
  unsigned int r = v.u + 0x7FFFu + ((v.u >> 16) & 1u);   // RNE
  return r >> 16;
}
__device__ __forceinline__ unsigned int pack2bf(float a, float b) {
  return f2bf1(a) | (f2bf1(b) << 16);
}
__device__ __forceinline__ float silu_f(float v) { return v / (1.0f + __expf(-v)); }

#if HAVE_TDM
// ---------------------------------------------------------------------------
// TDM: async-copy one bf16 tile [tile_d1 rows x BK cols] (row stride = K elems)
// from global to LDS; D# pads +16B after each 64B row -> LDS row stride 80B.
// Rows beyond tens_d1 read as zero (OOB rule) -> free zero-fill for small N.
// ---------------------------------------------------------------------------
__device__ __forceinline__ void tdm_load_tile_bf16(
    unsigned lds_off, const unsigned short* gsrc,
    int tile_d0, int tile_d1, int tens_d1, long stride_elems)
{
  unsigned long long ga = (unsigned long long)(const void*)gsrc;
  u32x4 g0;
  g0[0] = 1u;                                             // count=1, user mode
  g0[1] = lds_off;                                        // lds_addr
  g0[2] = (unsigned)(ga & 0xFFFFFFFFull);                 // global_addr[31:0]
  g0[3] = (unsigned)((ga >> 32) & 0x01FFFFFFull) | (2u << 30); // [56:32] + type=2
  const unsigned w0 = (1u << 16)    // data_size = 2 bytes
                    | (1u << 20)    // pad_enable
                    | (3u << 22)    // pad_interval: 16 DWORDs (64B)
                    | (3u << 25);   // pad_amount:   4 DWORDs (16B)
  unsigned long long q0 = (unsigned long long)w0
      | ((unsigned long long)(unsigned)(tile_d0 & 0xFFFF) << 48);  // tensor_dim0 lo16
  unsigned long long q1 =
        ((unsigned long long)(unsigned)tens_d1 << 16)              // tensor_dim1
      | ((unsigned long long)(unsigned)(tile_d0 & 0xFFFF) << 48);  // tile_dim0
  unsigned long long q2 = (unsigned long long)(unsigned)(tile_d1 & 0xFFFF)   // tile_dim1
      | (((unsigned long long)stride_elems & 0xFFFFFFFFull) << 32);          // stride lo
  unsigned long long q3 = ((unsigned long long)stride_elems >> 32) & 0xFFFFull;
  i32x8 g1;
  g1[0] = (int)(unsigned)(q0 & 0xFFFFFFFFull); g1[1] = (int)(unsigned)(q0 >> 32);
  g1[2] = (int)(unsigned)(q1 & 0xFFFFFFFFull); g1[3] = (int)(unsigned)(q1 >> 32);
  g1[4] = (int)(unsigned)(q2 & 0xFFFFFFFFull); g1[5] = (int)(unsigned)(q2 >> 32);
  g1[6] = (int)(unsigned)(q3 & 0xFFFFFFFFull); g1[7] = (int)(unsigned)(q3 >> 32);
  i32x4 gz;  gz[0] = 0; gz[1] = 0; gz[2] = 0; gz[3] = 0;
  i32x8 gz8; for (int q = 0; q < 8; q++) gz8[q] = 0;
  // 6-arg variant (clang-23 / therock): (g0, g1, g2, g3, g4, cpol)
  __builtin_amdgcn_tensor_load_to_lds(g0, g1, gz, gz, gz8, 0);
}
#endif

// ---------------------------------------------------------------------------
// WMMA GEMM: C[M,N] = A[M,K] * W[N,K]^T,  bf16 compute / f32 accumulate.
//   AMODE_PLAIN : A is bf16 in memory, staged via TDM (double-buffered LDS)
//   AMODE_IM2COL: A rows = 16x16x3 patches gathered from f32 x[32,3,256,256]
//   AMODE_KAN   : A[row, i*8+k] = spline basis_k(h[row,i]) (k<5), else 0
//   B (weights) always bf16, staged via TDM.
// ---------------------------------------------------------------------------
template<int AMODE, int EPI>
__global__ __launch_bounds__(256)
void gemm_wmma(const void* __restrict__ Aop, const void* __restrict__ Wop,
               float* __restrict__ C, unsigned short* __restrict__ Cbf,
               int M, int N, int K,
               const float* __restrict__ bias,
               const float* __restrict__ aux1, const float* __restrict__ aux2)
{
  __shared__ __align__(16) unsigned short As[2][BM * BKP];
  __shared__ __align__(16) unsigned short Bs[2][BN * BKP];

  const int tid  = threadIdx.x;
  const int lane = tid & 31;
  const int wave = tid >> 5;
  const int m0 = blockIdx.y * BM;
  const int n0 = blockIdx.x * BN;
  const int mW = (wave & 3) * 32;
  const int nW = (wave >> 2) * 32;
  const int arow = tid >> 1,  acol = (tid & 1) * 16;
  const int brow = tid >> 2,  bcol = (tid & 3) * 8;
  const int lhalf = lane >> 4, l15 = lane & 15;
  const int nk = K / BK;
  int bRows = N - n0; if (bRows > BN) bRows = BN;

  const unsigned short* Abf = (const unsigned short*)Aop;
  const unsigned short* Wbf = (const unsigned short*)Wop;
  const float*          Af  = (const float*)Aop;

  v8f acc[2][2];
  for (int i = 0; i < 2; i++)
    for (int j = 0; j < 2; j++)
      for (int q = 0; q < 8; q++) acc[i][j][q] = 0.0f;

  union Frag { u32x4 u[2]; v16bf v; };

  // A-operand generation for the custom modes (16 elems per thread per tile)
  auto stageA_regs = [&](int kb, float av[16]) {
    if constexpr (AMODE == AMODE_IM2COL) {
      const int grow = m0 + arow;
      const int b  = grow >> 8, pp = grow & 255;
      const int py = pp >> 4,  px = pp & 15;
      const int kbb = kb + acol;
      const int cch = kbb >> 8, rem = kbb & 255;
      const int iy  = rem >> 4;
      const float* p = Af + (((size_t)(b * 3 + cch) * 256 + py * 16 + iy) * 256 + px * 16);
      #pragma unroll
      for (int q = 0; q < 4; q++) {
        f32x4 t = ((const f32x4*)p)[q];
        av[4*q+0] = t[0]; av[4*q+1] = t[1]; av[4*q+2] = t[2]; av[4*q+3] = t[3];
      }
    } else if constexpr (AMODE == AMODE_KAN) {
      const float* hrow = Af + (size_t)(m0 + arow) * DIMC;
      #pragma unroll
      for (int e = 0; e < 16; e++) {
        const int kk = kb + acol + e;
        const int ii = kk >> 3, k5 = kk & 7;
        float val = 0.0f;
        if (k5 < 5) {
          const float xv = hrow[ii];
          const float ts = (xv + 1.0f - 0.4f * (float)k5) * 2.4999999f;
          val = (ts >= 0.0f && ts < 1.0f) ? ts : 0.0f;
        }
        av[e] = val;
      }
    }
  };
  auto writeA = [&](unsigned short* buf, const float av[16]) {
    unsigned int* da = (unsigned int*)&buf[arow * BKP + acol];
    #pragma unroll
    for (int e = 0; e < 8; e++) da[e] = pack2bf(av[2*e], av[2*e+1]);
  };

#if HAVE_TDM
  const int wuni = __builtin_amdgcn_readfirstlane(wave);
#endif

  // -------- prologue: stage k-tile 0 --------
  if constexpr (AMODE != AMODE_PLAIN) {
    float av[16];
    stageA_regs(0, av);
    writeA(&As[0][0], av);
  }
#if HAVE_TDM
  if (wuni == 0) {
    if constexpr (AMODE == AMODE_PLAIN)
      tdm_load_tile_bf16((unsigned)(size_t)(const void*)&As[0][0],
                         Abf + (size_t)m0 * K, BK, BM, BM, (long)K);
    tdm_load_tile_bf16((unsigned)(size_t)(const void*)&Bs[0][0],
                       Wbf + (size_t)n0 * K, BK, BN, bRows, (long)K);
    __builtin_amdgcn_s_wait_tensorcnt(0);
  }
#else
  if constexpr (AMODE == AMODE_PLAIN) {
    const u32x4* pa = (const u32x4*)(Abf + (size_t)(m0 + arow) * K + acol);
    u32x4* da = (u32x4*)&As[0][arow * BKP + acol];
    da[0] = pa[0]; da[1] = pa[1];
  }
  {
    const int gn = n0 + brow;
    u32x4 v; v[0] = 0; v[1] = 0; v[2] = 0; v[3] = 0;
    if (gn < N) v = *(const u32x4*)(Wbf + (size_t)gn * K + bcol);
    *(u32x4*)&Bs[0][brow * BKP + bcol] = v;
  }
#endif
  __syncthreads();

  // -------- main loop: double-buffered tiles --------
  for (int kt = 0; kt < nk; kt++) {
    const int cur = kt & 1, nxt = cur ^ 1;
    const int k1 = (kt + 1) * BK;
    const bool more = (kt + 1 < nk);

#if HAVE_TDM
    if (wuni == 0 && more) {
      if constexpr (AMODE == AMODE_PLAIN)
        tdm_load_tile_bf16((unsigned)(size_t)(const void*)&As[nxt][0],
                           Abf + (size_t)m0 * K + k1, BK, BM, BM, (long)K);
      tdm_load_tile_bf16((unsigned)(size_t)(const void*)&Bs[nxt][0],
                         Wbf + (size_t)n0 * K + k1, BK, BN, bRows, (long)K);
    }
#endif
    float av[16];
    if constexpr (AMODE != AMODE_PLAIN) { if (more) stageA_regs(k1, av); }
#if !HAVE_TDM
    u32x4 na0, na1, nb;
    if (more) {
      if constexpr (AMODE == AMODE_PLAIN) {
        const u32x4* pa = (const u32x4*)(Abf + (size_t)(m0 + arow) * K + k1 + acol);
        na0 = pa[0]; na1 = pa[1];
      }
      const int gn = n0 + brow;
      nb[0] = 0; nb[1] = 0; nb[2] = 0; nb[3] = 0;
      if (gn < N) nb = *(const u32x4*)(Wbf + (size_t)gn * K + k1 + bcol);
    }
#endif

    // fragments per documented 16-bit VGPR layouts (from buffers[cur])
    Frag fa[2], fb[2];
    #pragma unroll
    for (int i = 0; i < 2; i++) {
      const u32x4* q = (const u32x4*)&As[cur][(mW + 16*i + l15) * BKP];
      fa[i].u[0] = q[lhalf];
      fa[i].u[1] = q[2 + lhalf];
    }
    #pragma unroll
    for (int j = 0; j < 2; j++) {
      const u32x4* q = (const u32x4*)&Bs[cur][(nW + 16*j + l15) * BKP];
      fb[j].u[0] = q[lhalf];
      fb[j].u[1] = q[2 + lhalf];
    }
    #pragma unroll
    for (int i = 0; i < 2; i++)
      #pragma unroll
      for (int j = 0; j < 2; j++)
        acc[i][j] = __builtin_amdgcn_wmma_f32_16x16x32_bf16(
            false, fa[i].v, false, fb[j].v, (short)0, acc[i][j], false, false);

    if constexpr (AMODE != AMODE_PLAIN) { if (more) writeA(&As[nxt][0], av); }
#if !HAVE_TDM
    if (more) {
      if constexpr (AMODE == AMODE_PLAIN) {
        u32x4* da = (u32x4*)&As[nxt][arow * BKP + acol];
        da[0] = na0; da[1] = na1;
      }
      *(u32x4*)&Bs[nxt][brow * BKP + bcol] = nb;
    }
    __syncthreads();
#else
    __syncthreads();                                   // done reading buffers[cur]
    if (wuni == 0 && more) __builtin_amdgcn_s_wait_tensorcnt(0);
    __syncthreads();                                   // buffers[nxt] published
#endif
  }

  // -------- epilogue: C/D layout (m = r + 8*(lane>=16), n = lane&15) --------
  const int mloc = lhalf * 8;
  #pragma unroll
  for (int i = 0; i < 2; i++) {
    #pragma unroll
    for (int j = 0; j < 2; j++) {
      const int col = n0 + nW + 16*j + l15;
      if (col < N) {
        #pragma unroll
        for (int r = 0; r < 8; r++) {
          const int row = m0 + mW + 16*i + mloc + r;
          float v = acc[i][j][r];
          if (EPI == EPI_BIAS)          v += bias[col];
          else if (EPI == EPI_BIAS_POS) v += bias[col] + aux1[(size_t)(row & 255) * DIMC + col];
          else if (EPI == EPI_SOFTPLUS_BIAS) {
            v += bias[col];
            v = (v > 20.0f) ? v : log1pf(__expf(v));
          }
          else if (EPI == EPI_ADD2)     v += aux1[(size_t)row * N + col] + aux2[(size_t)row * N + col];
          C[(size_t)row * N + col] = v;
          if (Cbf) Cbf[(size_t)row * N + col] = (unsigned short)f2bf1(v);
        }
      }
    }
  }
}

// ---------------- f32 -> bf16 bulk conversion (8 elems/thread) ----------------
__global__ void cvt_bf16_k(const float* __restrict__ src, unsigned int* __restrict__ dst, long n8)
{
  const long i = (long)blockIdx.x * blockDim.x + threadIdx.x;
  if (i >= n8) return;
  const f32x4* s = (const f32x4*)src + i * 2;
  f32x4 a = s[0], b = s[1];
  dst[i*4+0] = pack2bf(a[0], a[1]);
  dst[i*4+1] = pack2bf(a[2], a[3]);
  dst[i*4+2] = pack2bf(b[0], b[1]);
  dst[i*4+3] = pack2bf(b[2], b[3]);
}

// ---------------- causal depthwise conv (DC=4) + SiLU ----------------
__global__ void conv_silu_k(const float* __restrict__ xr, const float* __restrict__ cw,
                            const float* __restrict__ cb, float* __restrict__ x1,
                            unsigned short* __restrict__ x1bf)
{
  const int idx = blockIdx.x * blockDim.x + threadIdx.x;
  if (idx >= TOK * DIC) return;
  const int ch = idx % DIC;
  const int t  = idx / DIC;
  const int l  = t & (SEQ - 1);
  float a = cb[ch];
  const float* base = xr + (size_t)t * (2 * DIC) + ch;
  if (l >= 3) a += cw[ch*4+0] * base[-3 * (2*DIC)];
  if (l >= 2) a += cw[ch*4+1] * base[-2 * (2*DIC)];
  if (l >= 1) a += cw[ch*4+2] * base[-1 * (2*DIC)];
  a += cw[ch*4+3] * base[0];
  const float s = silu_f(a);
  x1[idx]   = s;
  x1bf[idx] = (unsigned short)f2bf1(s);
}

// ---------------- u[t] = mean(x1[t, 0:96]) ----------------
__global__ void mean_u_k(const float* __restrict__ x1, float* __restrict__ u)
{
  const int t = blockIdx.x * blockDim.x + threadIdx.x;
  if (t >= TOK) return;
  const float* row = x1 + (size_t)t * DIC;
  float s = 0.0f;
  #pragma unroll
  for (int q = 0; q < 24; q++) {
    f32x4 v = ((const f32x4*)row)[q];
    s += v[0] + v[1] + v[2] + v[3];
  }
  u[t] = s * (1.0f / 96.0f);
}

// ---------------- Ad = exp(-exp(A_log)*dt),  Bd = dt*Bp*u ----------------
__global__ void scan_prep_k(const float* __restrict__ dt, const float* __restrict__ xdbl,
                            const float* __restrict__ u, const float* __restrict__ alog,
                            float* __restrict__ Ad, float* __restrict__ Bd)
{
  const int idx = blockIdx.x * blockDim.x + threadIdx.x;
  if (idx >= TOK * DSC) return;
  const int n = idx & 15;
  const int t = idx >> 4;
  const float dtv = dt[idx];
  Ad[idx] = __expf(-__expf(alog[n]) * dtv);
  Bd[idx] = dtv * xdbl[t * 32 + 16 + n] * u[t];
}

// ---------------- sequential scan: 32 batches x 16 states ----------------
__global__ void scan_k(const float* __restrict__ Ad, const float* __restrict__ Bd,
                       float* __restrict__ ys)
{
  const int c = blockIdx.x * blockDim.x + threadIdx.x;
  if (c >= 32 * DSC) return;
  const int b = c >> 4, n = c & 15;
  float s = 0.0f;
  const size_t base = (size_t)b * SEQ * DSC + n;
  for (int l = 0; l < SEQ; l++) {
    const size_t id = base + (size_t)l * DSC;
    s = s * Ad[id] + Bd[id];
    ys[id] = s;
  }
}

// ---------------- y = (repeat(ys,96) + x1*D) * silu(res) ----------------
__global__ void gate_k(const float* __restrict__ ys, const float* __restrict__ x1,
                       const float* __restrict__ xr, const float* __restrict__ Dp,
                       float* __restrict__ y, unsigned short* __restrict__ ybf)
{
  const int idx = blockIdx.x * blockDim.x + threadIdx.x;
  if (idx >= TOK * DIC) return;
  const int ch = idx % DIC;
  const int t  = idx / DIC;
  const float v = ys[t * DSC + ch / 96] + x1[idx] * Dp[ch];
  const float r = xr[(size_t)t * (2 * DIC) + DIC + ch];
  const float o = v * silu_f(r);
  y[idx]   = o;
  ybf[idx] = (unsigned short)f2bf1(o);
}

// ---------------- final LayerNorm ----------------
__global__ __launch_bounds__(256)
void layernorm_k(const float* __restrict__ h, const float* __restrict__ w,
                 const float* __restrict__ b, float* __restrict__ out)
{
  const int t = blockIdx.x;
  const int tid = threadIdx.x;
  __shared__ float red[256];
  const float* row = h + (size_t)t * DIMC;
  const float v0 = row[tid], v1 = row[tid + 256], v2 = row[tid + 512];
  red[tid] = v0 + v1 + v2;
  __syncthreads();
  for (int st = 128; st > 0; st >>= 1) { if (tid < st) red[tid] += red[tid + st]; __syncthreads(); }
  const float mu = red[0] * (1.0f / 768.0f);
  __syncthreads();
  const float d0 = v0 - mu, d1 = v1 - mu, d2 = v2 - mu;
  red[tid] = d0*d0 + d1*d1 + d2*d2;
  __syncthreads();
  for (int st = 128; st > 0; st >>= 1) { if (tid < st) red[tid] += red[tid + st]; __syncthreads(); }
  const float rstd = rsqrtf(red[0] * (1.0f / 768.0f) + 1e-5f);
  out[(size_t)t * DIMC + tid      ] = d0 * rstd * w[tid      ] + b[tid      ];
  out[(size_t)t * DIMC + tid + 256] = d1 * rstd * w[tid + 256] + b[tid + 256];
  out[(size_t)t * DIMC + tid + 512] = d2 * rstd * w[tid + 512] + b[tid + 512];
}

// ===========================================================================
extern "C" void kernel_launch(void* const* d_in, const int* in_sizes, int n_in,
                              void* d_out, int out_size, void* d_ws, size_t ws_size,
                              hipStream_t stream)
{
  (void)in_sizes; (void)n_in; (void)out_size; (void)ws_size;
  const float* x       = (const float*)d_in[0];
  const float* patch_w = (const float*)d_in[1];
  const float* patch_b = (const float*)d_in[2];
  const float* pos     = (const float*)d_in[3];
  const float* in_w    = (const float*)d_in[4];
  const float* conv_w  = (const float*)d_in[5];
  const float* conv_b  = (const float*)d_in[6];
  const float* xp_w    = (const float*)d_in[7];
  const float* dt_w    = (const float*)d_in[8];
  const float* dt_b    = (const float*)d_in[9];
  const float* A_log   = (const float*)d_in[10];
  const float* D_par   = (const float*)d_in[11];
  const float* out_w   = (const float*)d_in[12];
  const float* kb_w    = (const float*)d_in[13];
  const float* k_coeff = (const float*)d_in[14];
  const float* k_bias  = (const float*)d_in[15];
  const float* norm_w  = (const float*)d_in[16];
  const float* norm_b  = (const float*)d_in[17];

  char* ws = (char*)d_ws;
  size_t off = 0;
  auto take = [&](size_t bytes) -> char* {
    char* p = ws + off;
    off += (bytes + 255) & ~(size_t)255;
    return p;
  };
  // f32 activations
  float* hA   = (float*)take((size_t)TOK * DIMC * 4);
  float* hB   = (float*)take((size_t)TOK * DIMC * 4);
  float* xr   = (float*)take((size_t)TOK * 2 * DIC * 4);   // also KAN base tmp
  float* x1   = (float*)take((size_t)TOK * DIC * 4);
  float* yb   = (float*)take((size_t)TOK * DIC * 4);
  float* xdbl = (float*)take((size_t)TOK * 32 * 4);
  float* dtb  = (float*)take((size_t)TOK * DSC * 4);
  float* Adb  = (float*)take((size_t)TOK * DSC * 4);
  float* Bdb  = (float*)take((size_t)TOK * DSC * 4);
  float* ysb  = (float*)take((size_t)TOK * DSC * 4);
  float* ub   = (float*)take((size_t)TOK * 4);
  float* kanb = xr;
  // bf16 shadows (GEMM operands)
  unsigned short* h_bf  = (unsigned short*)take((size_t)TOK * DIMC * 2);
  unsigned short* x1_bf = (unsigned short*)take((size_t)TOK * DIC * 2);
  unsigned short* y_bf  = (unsigned short*)take((size_t)TOK * DIC * 2);
  unsigned short* w_in  = (unsigned short*)take((size_t)NLAY * 2 * DIC * DIMC * 2);
  unsigned short* w_xp  = (unsigned short*)take((size_t)NLAY * 32 * DIC * 2);
  unsigned short* w_dt  = (unsigned short*)take((size_t)NLAY * DSC * DIC * 2);
  unsigned short* w_out = (unsigned short*)take((size_t)NLAY * DIMC * DIC * 2);
  unsigned short* w_kb  = (unsigned short*)take((size_t)4 * DIMC * DIMC * 2);
  unsigned short* w_kc  = (unsigned short*)take((size_t)4 * DIMC * KANK * 2);
  unsigned short* w_pt  = (unsigned short*)take((size_t)DIMC * DIMC * 2);

  const dim3 blk(256);
  auto grid1 = [](long total) { return dim3((unsigned)((total + 255) / 256)); };
  auto cvt = [&](const float* src, unsigned short* dst, long n) {
    const long n8 = n / 8;
    cvt_bf16_k<<<grid1(n8), blk, 0, stream>>>(src, (unsigned int*)dst, n8);
  };

  // ---- one-time (per launch) weight conversion to bf16 ----
  cvt(in_w,    w_in,  (long)NLAY * 2 * DIC * DIMC);
  cvt(xp_w,    w_xp,  (long)NLAY * 32 * DIC);
  cvt(dt_w,    w_dt,  (long)NLAY * DSC * DIC);
  cvt(out_w,   w_out, (long)NLAY * DIMC * DIC);
  cvt(kb_w,    w_kb,  (long)4 * DIMC * DIMC);
  cvt(k_coeff, w_kc,  (long)4 * DIMC * KANK);
  cvt(patch_w, w_pt,  (long)DIMC * DIMC);

  // ---- patch embed: im2col GEMM + bias + pos_embed -> hA (+ h_bf) ----
  {
    dim3 g((DIMC + BN - 1) / BN, TOK / BM);
    gemm_wmma<AMODE_IM2COL, EPI_BIAS_POS><<<g, blk, 0, stream>>>(
        x, w_pt, hA, h_bf, TOK, DIMC, DIMC, patch_b, pos, nullptr);
  }

  float* cur = hA;
  float* alt = hB;

  for (int i = 0; i < NLAY; i++) {
    {   // in_proj: [TOK,768] x [3072,768]^T -> xr
      dim3 g((2 * DIC + BN - 1) / BN, TOK / BM);
      gemm_wmma<AMODE_PLAIN, EPI_NONE><<<g, blk, 0, stream>>>(
          h_bf, w_in + (size_t)i * 2 * DIC * DIMC, xr, nullptr,
          TOK, 2 * DIC, DIMC, nullptr, nullptr, nullptr);
    }
    conv_silu_k<<<grid1((long)TOK * DIC), blk, 0, stream>>>(
        xr, conv_w + (size_t)i * DIC * 4, conv_b + (size_t)i * DIC, x1, x1_bf);
    {   // x_proj -> xdbl (N=32, B-tile zero-filled by TDM OOB rule)
      dim3 g(1, TOK / BM);
      gemm_wmma<AMODE_PLAIN, EPI_NONE><<<g, blk, 0, stream>>>(
          x1_bf, w_xp + (size_t)i * 32 * DIC, xdbl, nullptr,
          TOK, 32, DIC, nullptr, nullptr, nullptr);
    }
    {   // dt -> softplus(. + dt_b)   (N=16)
      dim3 g(1, TOK / BM);
      gemm_wmma<AMODE_PLAIN, EPI_SOFTPLUS_BIAS><<<g, blk, 0, stream>>>(
          x1_bf, w_dt + (size_t)i * DSC * DIC, dtb, nullptr,
          TOK, DSC, DIC, dt_b + (size_t)i * DSC, nullptr, nullptr);
    }
    mean_u_k<<<grid1(TOK), blk, 0, stream>>>(x1, ub);
    scan_prep_k<<<grid1((long)TOK * DSC), blk, 0, stream>>>(
        dtb, xdbl, ub, A_log + (size_t)i * DSC, Adb, Bdb);
    scan_k<<<dim3(2), blk, 0, stream>>>(Adb, Bdb, ysb);
    gate_k<<<grid1((long)TOK * DIC), blk, 0, stream>>>(
        ysb, x1, xr, D_par + (size_t)i * DIC, yb, y_bf);
    {   // out_proj -> cur (+ h_bf)
      dim3 g((DIMC + BN - 1) / BN, TOK / BM);
      gemm_wmma<AMODE_PLAIN, EPI_NONE><<<g, blk, 0, stream>>>(
          y_bf, w_out + (size_t)i * DIMC * DIC, cur, h_bf,
          TOK, DIMC, DIC, nullptr, nullptr, nullptr);
    }
    if (i % 3 == 2) {   // KAN: h = base + spline + h
      const int j = i / 3;
      {
        dim3 g((DIMC + BN - 1) / BN, TOK / BM);
        gemm_wmma<AMODE_PLAIN, EPI_BIAS><<<g, blk, 0, stream>>>(
            h_bf, w_kb + (size_t)j * DIMC * DIMC, kanb, nullptr,
            TOK, DIMC, DIMC, k_bias + (size_t)j * DIMC, nullptr, nullptr);
      }
      {
        dim3 g((DIMC + BN - 1) / BN, TOK / BM);
        gemm_wmma<AMODE_KAN, EPI_ADD2><<<g, blk, 0, stream>>>(
            cur, w_kc + (size_t)j * DIMC * KANK, alt, h_bf,
            TOK, DIMC, KANK, nullptr, kanb, cur);
      }
      float* t = cur; cur = alt; alt = t;
    }
  }

  layernorm_k<<<dim3(TOK), blk, 0, stream>>>(cur, norm_w, norm_b, (float*)d_out);
}
